// MEGNet_7499012898896
// MI455X (gfx1250) — matrix operand order repaired and checked by reading, not compiled, not run on hardware
//
#include <hip/hip_runtime.h>
#include <hip/hip_bf16.h>
#include <cstddef>
#include <cstdint>

// ---------------------------------------------------------------------------
// WMMA GEMM core: C[M,Ntot] = act(A[M,K] @ W[K,Ntot] + b1 + b2)
// A is described by up to 4 concatenated segments, each optionally row-gathered.
// bf16 inputs staged into LDS in *fragment order* (k = 16*(lane>>4) + e), f32 accumulate
// via v_wmma_f32_16x16x32_bf16. Fragment loads are single 32B LDS vector
// reads (2x ds_load_b128 per fragment). b1/b2 must be valid pointers (host
// passes a zero buffer when a bias is absent) -> branch-free epilogue.
// ---------------------------------------------------------------------------

#define BM 128
#define BK 32
#define MAXCH 10          // supports K up to 320
#define SA_STRIDE 40      // bf16 elems per A row (32 data + 8 pad; 80B, 16B-aligned)
#define BN_GRID 512

typedef __attribute__((ext_vector_type(16))) __bf16 v16bf_t;
typedef __attribute__((ext_vector_type(8)))  float  v8f_t;

struct ASrc {
    const float* ptr[4];
    const int*   idx[4];   // nullptr => identity row
    int          ld[4];    // row stride of source (== segment width here)
    int          kend[4];  // cumulative column end
    int          nseg;
};

__device__ __forceinline__ unsigned short f2bf(float f) {
    unsigned int u = __float_as_uint(f);
    u += 0x7FFFu + ((u >> 16) & 1u);   // round-to-nearest-even
    return (unsigned short)(u >> 16);
}

__device__ __forceinline__ const float* loadAp(const ASrc& A, int r, int k) {
    int s = 0, k0 = 0;
    while (s + 1 < A.nseg && k >= A.kend[s]) { k0 = A.kend[s]; ++s; }
    const int* ix = A.idx[s];
    int rr = ix ? ix[r] : r;
    return &A.ptr[s][(size_t)rr * A.ld[s] + (k - k0)];
}

// Fragment-order LDS layouts:
//  A: sAf[row*SA_STRIDE + hi*16 + e], lane(m,hi) reads 16 contiguous bf16.
//  W: sWf[(((chunk*4+j)*32 + lane)*16) + e], lane reads 16 contiguous bf16,
//     K = chunk*32 + (lane>>4)*16 + e, col = j*16 + (lane&15).
template <int CH>
__global__ __launch_bounds__(256)
void gemm_wmma(ASrc A, const float* __restrict__ W, const float* __restrict__ b1,
               const float* __restrict__ b2, float* __restrict__ C,
               int M, int K, int Nw, int ldC, int relu)
{
    __shared__ alignas(32) unsigned short sWf[MAXCH * 4 * 32 * 16];  // 40 KB
    __shared__ alignas(32) unsigned short sAf[BM * SA_STRIDE];       // 10 KB

    const int tid     = threadIdx.x;
    const int lane    = tid & 31;
    const int wid     = tid >> 5;
    const int rowBase = blockIdx.x * BM;
    const int colBase = blockIdx.y * 64;
    const int chunks  = (CH > 0) ? CH : ((K + BK - 1) / BK);

    const int n  = lane & 15;
    const int hi = lane >> 4;

    // ---- bias loads issued early; latency hides under the WMMA loop ----
    const float* bp1 = b1 + colBase + n;
    const float* bp2 = b2 + colBase + n;
    float bias[4];
#pragma unroll
    for (int j = 0; j < 4; ++j) bias[j] = bp1[j * 16] + bp2[j * 16];

    // ---- stage weights, swizzled to fragment order, packed 2 bf16 / store ----
    for (int i = tid; i < chunks * 1024; i += 256) {   // chunks*4*32*8 pairs
        int ep   = i & 7;
        int ln   = (i >> 3) & 31;
        int j    = (i >> 8) & 3;
        int ch   = i >> 10;
        int e    = ep * 2;
        int k    = ch * 32 + (ln >> 4) * 16 + e;
        int col  = colBase + j * 16 + (ln & 15);
        float v0 = (k     < K) ? W[(size_t)k       * Nw + col] : 0.f;
        float v1 = (k + 1 < K) ? W[(size_t)(k + 1) * Nw + col] : 0.f;
        unsigned int pk = (unsigned)f2bf(v0) | ((unsigned)f2bf(v1) << 16);
        *(unsigned int*)&sWf[(size_t)(((ch * 4 + j) * 32 + ln) * 16 + e)] = pk;
    }

    v8f_t acc[4] = {v8f_t{}, v8f_t{}, v8f_t{}, v8f_t{}};

    const int m    = lane & 15;
    const int arow = wid * 16 + m;

#pragma unroll
    for (int kc = 0; kc < chunks; ++kc) {
        const int k0 = kc * BK;
        __syncthreads();
        // ---- stage A tile (BM x BK) swizzled, fused gather/concat + prefetch ----
        for (int i = tid; i < (BM * BK) / 2; i += 256) {
            int rl = i >> 4;
            int kl = (i & 15) * 2;
            int r  = rowBase + rl, k = k0 + kl;
            float v0 = 0.f, v1 = 0.f;
            if (r < M && k < K) {
                const float* p = loadAp(A, r, k);   // segment widths are even, so
                v0 = p[0];                          // k and k+1 share a segment
                v1 = (k + 1 < K) ? p[1] : 0.f;
                if (k + BK < K) __builtin_prefetch(p + BK, 0, 1);  // next chunk
            }
            int h2 = kl >> 4;
            int e  = kl & 15;
            *(unsigned int*)&sAf[rl * SA_STRIDE + h2 * 16 + e] =
                (unsigned)f2bf(v0) | ((unsigned)f2bf(v1) << 16);
        }
        __syncthreads();

        // ---- fragments: one 32B LDS vector read each (2x ds_load_b128) ----
        v16bf_t af = *(const v16bf_t*)(const void*)&sAf[arow * SA_STRIDE + hi * 16];
#pragma unroll
        for (int j = 0; j < 4; ++j) {
            v16bf_t bfv = *(const v16bf_t*)(const void*)&sWf[(size_t)(((kc * 4 + j) * 32 + lane) * 16)];
            acc[j] = __builtin_amdgcn_wmma_f32_16x16x32_bf16(
                         false, af, false, bfv, (short)0, acc[j], false, false);
        }
    }

    // ---- epilogue: branch-free bias + uniform relu + strength-reduced stores ----
    const int mb = hi * 8;
#pragma unroll
    for (int j = 0; j < 4; ++j)
#pragma unroll
        for (int r = 0; r < 8; ++r) acc[j][r] += bias[j];
    if (relu) {
#pragma unroll
        for (int j = 0; j < 4; ++j)
#pragma unroll
            for (int r = 0; r < 8; ++r) acc[j][r] = fmaxf(acc[j][r], 0.f);
    }

    const int row0 = rowBase + wid * 16 + mb;
    float* Cp = C + (size_t)row0 * ldC + colBase + n;
    if (rowBase + BM <= M) {            // interior block: branch-free stores
#pragma unroll
        for (int r = 0; r < 8; ++r) {
#pragma unroll
            for (int j = 0; j < 4; ++j) Cp[j * 16] = acc[j][r];
            Cp += ldC;
        }
    } else {                            // boundary block: per-row guard only
#pragma unroll
        for (int r = 0; r < 8; ++r) {
            if (row0 + r < M) {
#pragma unroll
                for (int j = 0; j < 4; ++j) Cp[j * 16] = acc[j][r];
            }
            Cp += ldC;
        }
    }
}

// ---------------------------------------------------------------------------
// BatchNorm (training stats, biased variance) — deterministic 2-stage reduce.
// ---------------------------------------------------------------------------
__global__ void bn_partial(const float* __restrict__ x, int M, float* __restrict__ part) {
    __shared__ float ss[256], sq[256];
    int col = threadIdx.x & 63;
    int sub = threadIdx.x >> 6;
    float s = 0.f, q = 0.f;
    for (int r = blockIdx.x * 4 + sub; r < M; r += gridDim.x * 4) {
        float v = x[(size_t)r * 64 + col];
        s += v; q += v * v;
    }
    ss[threadIdx.x] = s; sq[threadIdx.x] = q;
    __syncthreads();
    if (threadIdx.x < 64) {
        float S = 0.f, Q = 0.f;
        for (int i = 0; i < 4; ++i) { S += ss[col + i * 64]; Q += sq[col + i * 64]; }
        part[blockIdx.x * 128 + col]      = S;
        part[blockIdx.x * 128 + 64 + col] = Q;
    }
}

__global__ void bn_finalize(const float* __restrict__ part, int G, int M, float* __restrict__ stats) {
    int c = threadIdx.x;  // 64 threads
    float S = 0.f, Q = 0.f;
    for (int g = 0; g < G; ++g) { S += part[g * 128 + c]; Q += part[g * 128 + 64 + c]; }
    float mInv = 1.f / (float)M;
    float mu = S * mInv;
    float var = Q * mInv - mu * mu;
    stats[c]      = mu;
    stats[64 + c] = rsqrtf(var + 1e-5f);
}

__global__ void bn_apply(float* __restrict__ x, size_t n, const float* __restrict__ stats,
                         const float* __restrict__ g, const float* __restrict__ be) {
    size_t i = (size_t)blockIdx.x * 256 + threadIdx.x;
    if (i >= n) return;
    int c = (int)(i & 63);
    x[i] = (x[i] - stats[c]) * stats[64 + c] * g[c] + be[c];
}

// ---------------------------------------------------------------------------
// Segment / elementwise helpers
// ---------------------------------------------------------------------------
__global__ void kfill0(float* __restrict__ p, size_t n) {
    size_t i = (size_t)blockIdx.x * 256 + threadIdx.x;
    if (i < n) p[i] = 0.f;
}

__global__ void gather_int(const int* __restrict__ row, const int* __restrict__ nb,
                           int E, int* __restrict__ eb) {
    int e = blockIdx.x * 256 + threadIdx.x;
    if (e < E) eb[e] = nb[row[e]];
}

__global__ void seg_scatter(const float* __restrict__ v, const int* __restrict__ row,
                            int E, float* __restrict__ acc, float* __restrict__ cnt) {
    size_t i = (size_t)blockIdx.x * 256 + threadIdx.x;
    if (i >= (size_t)E * 64) return;
    int e = (int)(i >> 6), c = (int)(i & 63);
    int r = row[e];
    atomicAdd(&acc[(size_t)r * 64 + c], v[i]);
    if (c == 0) atomicAdd(&cnt[r], 1.0f);
}

__global__ void seg_div(float* __restrict__ acc, const float* __restrict__ cnt, int N) {
    size_t i = (size_t)blockIdx.x * 256 + threadIdx.x;
    if (i >= (size_t)N * 64) return;
    acc[i] /= fmaxf(cnt[i >> 6], 1.0f);
}

// mean over contiguous per-graph segments (node_batch = repeat(arange(B), npg))
__global__ void graph_mean(const float* __restrict__ x, int npg, float* __restrict__ out) {
    int b = blockIdx.x, c = threadIdx.x;  // 64 threads
    const float* p = x + (size_t)b * npg * 64;
    float s = 0.f;
    for (int i = 0; i < npg; ++i) s += p[(size_t)i * 64 + c];
    out[(size_t)b * 64 + c] = s / (float)npg;
}

__global__ void add2(float* __restrict__ dst, const float* __restrict__ a,
                     const float* __restrict__ b, size_t n) {
    size_t i = (size_t)blockIdx.x * 256 + threadIdx.x;
    if (i < n) dst[i] = a[i] + b[i];
}

// ---------------------------------------------------------------------------
// Set2Set pieces
// ---------------------------------------------------------------------------
__global__ void build_s2s_w(const float* __restrict__ Wih, const float* __restrict__ Whh,
                            const float* __restrict__ bih, const float* __restrict__ bhh,
                            float* __restrict__ Wc, float* __restrict__ bc) {
    int i = blockIdx.x * 256 + threadIdx.x;
    if (i < 192 * 256) {
        int k = i / 256, n = i % 256;
        Wc[i] = (k < 128) ? Wih[n * 128 + k] : Whh[n * 64 + (k - 128)];
    }
    if (i < 256) bc[i] = bih[i] + bhh[i];
}

__device__ __forceinline__ float sigf(float x) { return 1.f / (1.f + expf(-x)); }

__global__ void lstm_cell(const float* __restrict__ gates, float* __restrict__ h,
                          float* __restrict__ c, float* __restrict__ qs) {
    int b = blockIdx.x, k = threadIdx.x;  // 64 threads
    size_t g0 = (size_t)b * 256 + k;
    float ig = sigf(gates[g0]);
    float fg = sigf(gates[g0 + 64]);
    float gg = tanhf(gates[g0 + 128]);
    float og = sigf(gates[g0 + 192]);
    size_t ix = (size_t)b * 64 + k;
    float cn = fg * c[ix] + ig * gg;
    c[ix] = cn;
    float hn = og * tanhf(cn);
    h[ix] = hn;
    qs[(size_t)b * 128 + k] = hn;  // q part of q_star
}

__global__ void s2s_attn(const float* __restrict__ xf, const float* __restrict__ h,
                         int npg, float* __restrict__ qs) {
    __shared__ float e[128];
    __shared__ float red[128];
    int b = blockIdx.x, t = threadIdx.x;  // 128 threads, npg <= 128
    const float* xb = xf + (size_t)b * npg * 64;
    const float* q  = h  + (size_t)b * 64;
    float ev = -1e30f;
    if (t < npg) {
        float s = 0.f;
        for (int c = 0; c < 64; ++c) s += xb[(size_t)t * 64 + c] * q[c];
        e[t] = s; ev = s;
    }
    red[t] = ev;
    __syncthreads();
    for (int o = 64; o > 0; o >>= 1) { if (t < o) red[t] = fmaxf(red[t], red[t + o]); __syncthreads(); }
    float mx = red[0];
    __syncthreads();
    float ex = 0.f;
    if (t < npg) { ex = expf(e[t] - mx); e[t] = ex; }
    red[t] = ex;
    __syncthreads();
    for (int o = 64; o > 0; o >>= 1) { if (t < o) red[t] += red[t + o]; __syncthreads(); }
    float s = red[0];
    if (t < 64) {
        float r = 0.f;
        for (int i = 0; i < npg; ++i) r += e[i] * xb[(size_t)i * 64 + t];
        qs[(size_t)b * 128 + 64 + t] = r / s;  // r part of q_star
    }
}

__global__ void final_out(const float* __restrict__ post, const float* __restrict__ W,
                          const float* __restrict__ b, float* __restrict__ out) {
    __shared__ float red[64];
    int g = blockIdx.x, t = threadIdx.x;
    red[t] = post[(size_t)g * 64 + t] * W[t];
    __syncthreads();
    for (int o = 32; o > 0; o >>= 1) { if (t < o) red[t] += red[t + o]; __syncthreads(); }
    if (t == 0) out[g] = red[0] + b[0];
}

// ---------------------------------------------------------------------------
// Host-side orchestration
// ---------------------------------------------------------------------------
struct Lin { const float* W; const float* b; };
struct Emb { Lin l1, l2; };
struct Blk { Lin l[3]; const float* g[3]; const float* be[3]; };
struct S2S { const float* Wih; const float* Whh; const float* bih; const float* bhh; };

static Lin getLin(void* const* d, int& p) { Lin L; L.W = (const float*)d[p++]; L.b = (const float*)d[p++]; return L; }
static Emb getEmb(void* const* d, int& p) { Emb e; e.l1 = getLin(d, p); e.l2 = getLin(d, p); return e; }
static Blk getBlk(void* const* d, int& p) {
    Blk b;
    for (int i = 0; i < 3; ++i) b.l[i]  = getLin(d, p);
    for (int i = 0; i < 3; ++i) b.g[i]  = (const float*)d[p++];
    for (int i = 0; i < 3; ++i) b.be[i] = (const float*)d[p++];
    return b;
}
static S2S getS2S(void* const* d, int& p) {
    S2S s;
    s.Wih = (const float*)d[p++]; s.Whh = (const float*)d[p++];
    s.bih = (const float*)d[p++]; s.bhh = (const float*)d[p++];
    return s;
}

static void addSeg(ASrc& a, const float* ptr, const int* idx, int w) {
    int s = a.nseg;
    a.ptr[s] = ptr; a.idx[s] = idx; a.ld[s] = w;
    a.kend[s] = (s ? a.kend[s - 1] : 0) + w;
    a.nseg = s + 1;
}
static ASrc plainA(const float* p, int w) { ASrc a{}; addSeg(a, p, nullptr, w); return a; }

static void launch_gemm(hipStream_t st, const ASrc& A, const float* W, const float* b1,
                        const float* b2, float* C, int M, int K, int Nt, int relu) {
    dim3 grid((unsigned)((M + BM - 1) / BM), (unsigned)(Nt / 64));
    int chunks = (K + BK - 1) / BK;
    switch (chunks) {
    case 1:  gemm_wmma<1><<<grid, 256, 0, st>>>(A, W, b1, b2, C, M, K, Nt, Nt, relu); break;
    case 2:  gemm_wmma<2><<<grid, 256, 0, st>>>(A, W, b1, b2, C, M, K, Nt, Nt, relu); break;
    case 3:  gemm_wmma<3><<<grid, 256, 0, st>>>(A, W, b1, b2, C, M, K, Nt, Nt, relu); break;
    case 6:  gemm_wmma<6><<<grid, 256, 0, st>>>(A, W, b1, b2, C, M, K, Nt, Nt, relu); break;
    case 8:  gemm_wmma<8><<<grid, 256, 0, st>>>(A, W, b1, b2, C, M, K, Nt, Nt, relu); break;
    case 10: gemm_wmma<10><<<grid, 256, 0, st>>>(A, W, b1, b2, C, M, K, Nt, Nt, relu); break;
    default: gemm_wmma<0><<<grid, 256, 0, st>>>(A, W, b1, b2, C, M, K, Nt, Nt, relu); break;
    }
}

static void run_bn(hipStream_t st, float* x, int M, const float* g, const float* be,
                   float* part, float* stats) {
    int G = (M + 3) / 4; if (G > BN_GRID) G = BN_GRID; if (G < 1) G = 1;
    bn_partial<<<(unsigned)G, 256, 0, st>>>(x, M, part);
    bn_finalize<<<1, 64, 0, st>>>(part, G, M, stats);
    size_t n = (size_t)M * 64;
    bn_apply<<<(unsigned)((n + 255) / 256), 256, 0, st>>>(x, n, stats, g, be);
}

static void run_mlp2(hipStream_t st, const ASrc& in, int K0, int M, const Emb& E_,
                     float* tmp, float* out, const float* zb) {
    launch_gemm(st, in, E_.l1.W, E_.l1.b, zb, tmp, M, K0, 64, 1);
    launch_gemm(st, plainA(tmp, 64), E_.l2.W, E_.l2.b, zb, out, M, 64, 64, 1);
}

static void run_block(hipStream_t st, const ASrc& in0, int K0, int M, const Blk& B_,
                      float* t1, float* t2, float* part, float* stats, const float* zb) {
    launch_gemm(st, in0, B_.l[0].W, B_.l[0].b, zb, t1, M, K0, 64, 1);
    run_bn(st, t1, M, B_.g[0], B_.be[0], part, stats);
    launch_gemm(st, plainA(t1, 64), B_.l[1].W, B_.l[1].b, zb, t2, M, 64, 64, 1);
    run_bn(st, t2, M, B_.g[1], B_.be[1], part, stats);
    launch_gemm(st, plainA(t2, 64), B_.l[2].W, B_.l[2].b, zb, t1, M, 64, 64, 1);
    run_bn(st, t1, M, B_.g[2], B_.be[2], part, stats);
    // result in t1
}

static void run_set2set(hipStream_t st, const float* xf, int npg, int Bg, const float* Wc,
                        const float* bc, float* h, float* c, float* qs, float* gates,
                        const float* zb) {
    size_t n64 = (size_t)Bg * 64, n128 = (size_t)Bg * 128;
    kfill0<<<(unsigned)((n64 + 255) / 256), 256, 0, st>>>(h, n64);
    kfill0<<<(unsigned)((n64 + 255) / 256), 256, 0, st>>>(c, n64);
    kfill0<<<(unsigned)((n128 + 255) / 256), 256, 0, st>>>(qs, n128);
    for (int s = 0; s < 3; ++s) {
        ASrc A{};
        addSeg(A, qs, nullptr, 128);
        addSeg(A, h, nullptr, 64);
        launch_gemm(st, A, Wc, bc, zb, gates, Bg, 192, 256, 0);
        lstm_cell<<<(unsigned)Bg, 64, 0, st>>>(gates, h, c, qs);
        s2s_attn<<<(unsigned)Bg, 128, 0, st>>>(xf, h, npg, qs);
    }
}

extern "C" void kernel_launch(void* const* d_in, const int* in_sizes, int n_in,
                              void* d_out, int out_size, void* d_ws, size_t ws_size,
                              hipStream_t stream) {
    (void)n_in; (void)ws_size;
    // ---- inputs ----
    const float* x_in  = (const float*)d_in[0];
    const int*   eidx  = (const int*)d_in[1];
    const float* eattr = (const float*)d_in[2];
    const float* gattr = (const float*)d_in[3];
    const int*   nbat  = (const int*)d_in[4];

    const int N  = in_sizes[4];
    const int E  = in_sizes[1] / 2;
    const int Bg = out_size;
    const int Dx = in_sizes[0] / N;
    const int De = in_sizes[2] / E;
    const int Du = in_sizes[3] / Bg;
    const int npg = N / Bg;

    const int* row = eidx;
    const int* col = eidx + E;

    // ---- params (insertion order of setup_inputs) ----
    int p = 5;
    Lin pre = getLin(d_in, p);
    Emb embE[3], embN[3], embG[3];
    for (int i = 0; i < 3; ++i) { embE[i] = getEmb(d_in, p); embN[i] = getEmb(d_in, p); embG[i] = getEmb(d_in, p); }
    Blk convE[3], convN[3], convG[3];
    for (int i = 0; i < 3; ++i) { convE[i] = getBlk(d_in, p); convN[i] = getBlk(d_in, p); convG[i] = getBlk(d_in, p); }
    S2S s2sx = getS2S(d_in, p);
    S2S s2se = getS2S(d_in, p);
    Lin post = getLin(d_in, p);
    Lin fin  = getLin(d_in, p);

    // ---- workspace layout ----
    char* base = (char*)d_ws;
    size_t off = 0;
    auto alloc = [&](size_t bytes) -> void* {
        void* q = base + off;
        off += (bytes + 255) & ~(size_t)255;
        return q;
    };
    const size_t N64 = (size_t)N * 64 * 4, E64 = (size_t)E * 64 * 4, B64 = (size_t)Bg * 64 * 4;
    float* xc   = (float*)alloc(N64);
    float* x_t  = (float*)alloc(N64);
    float* xt1  = (float*)alloc(N64);
    float* xt2  = (float*)alloc(N64);
    float* ec   = (float*)alloc(E64);
    float* e_t  = (float*)alloc(E64);
    float* et1  = (float*)alloc(E64);
    float* et2  = (float*)alloc(E64);
    float* uc   = (float*)alloc(B64);
    float* u_t  = (float*)alloc(B64);
    float* ut1  = (float*)alloc(B64);
    float* ut2  = (float*)alloc(B64);
    float* u_e  = (float*)alloc(B64);
    float* u_v  = (float*)alloc(B64);
    float* segn = (float*)alloc(N64);
    float* cntn = (float*)alloc((size_t)N * 4);
    int*   eb   = (int*)alloc((size_t)E * 4);
    float* part = (float*)alloc((size_t)BN_GRID * 128 * 4);
    float* stats = (float*)alloc(128 * 4);
    float* WcX = (float*)alloc(192 * 256 * 4);
    float* bcX = (float*)alloc(256 * 4);
    float* WcE = (float*)alloc(192 * 256 * 4);
    float* bcE = (float*)alloc(256 * 4);
    float* hS  = (float*)alloc(B64);
    float* cS  = (float*)alloc(B64);
    float* gates = (float*)alloc((size_t)Bg * 256 * 4);
    float* qsx = (float*)alloc((size_t)Bg * 128 * 4);
    float* qse = (float*)alloc((size_t)Bg * 128 * 4);
    float* pout = (float*)alloc(B64);
    float* zeroB = (float*)alloc(256 * 4);   // always-valid zero bias

    // ---- precompute ----
    kfill0<<<1, 256, 0, stream>>>(zeroB, (size_t)256);
    gather_int<<<(unsigned)((E + 255) / 256), 256, 0, stream>>>(row, nbat, E, eb);
    build_s2s_w<<<192, 256, 0, stream>>>(s2sx.Wih, s2sx.Whh, s2sx.bih, s2sx.bhh, WcX, bcX);
    build_s2s_w<<<192, 256, 0, stream>>>(s2se.Wih, s2se.Whh, s2se.bih, s2se.bhh, WcE, bcE);

    // ---- pre: xc = relu(x @ W + b) ----
    launch_gemm(stream, plainA(x_in, Dx), pre.W, pre.b, zeroB, xc, N, Dx, 64, 1);

    const float* e_in = eattr; int Ke = De;
    const float* u_in = gattr; int Ku = Du;

    for (int i = 0; i < 3; ++i) {
        // embeddings (MLP2s)
        run_mlp2(stream, plainA(e_in, Ke), Ke, E, embE[i], et1, e_t, zeroB);
        run_mlp2(stream, plainA(xc, 64), 64, N, embN[i], xt2, x_t, zeroB);
        run_mlp2(stream, plainA(u_in, Ku), Ku, Bg, embG[i], ut2, u_t, zeroB);

        // edge block: concat[x_t[row], x_t[col], e_t, u_t[eb]]  (fused gather)
        ASrc Ae{};
        addSeg(Ae, x_t, row, 64);
        addSeg(Ae, x_t, col, 64);
        addSeg(Ae, e_t, nullptr, 64);
        addSeg(Ae, u_t, eb, 64);
        run_block(stream, Ae, 256, E, convE[i], et1, et2, part, stats, zeroB);  // e_out in et1

        // seg_mean(e_out, row, N)
        size_t nN64 = (size_t)N * 64;
        kfill0<<<(unsigned)((nN64 + 255) / 256), 256, 0, stream>>>(segn, nN64);
        kfill0<<<(unsigned)((N + 255) / 256), 256, 0, stream>>>(cntn, (size_t)N);
        seg_scatter<<<(unsigned)(((size_t)E * 64 + 255) / 256), 256, 0, stream>>>(et1, row, E, segn, cntn);
        seg_div<<<(unsigned)((nN64 + 255) / 256), 256, 0, stream>>>(segn, cntn, N);

        // node block: concat[x_t, segn, u_t[node_batch]]
        ASrc An{};
        addSeg(An, x_t, nullptr, 64);
        addSeg(An, segn, nullptr, 64);
        addSeg(An, u_t, nbat, 64);
        run_block(stream, An, 192, N, convN[i], xt1, xt2, part, stats, zeroB);  // x_out in xt1

        // global block: concat[u_e, u_v, u_t]
        graph_mean<<<(unsigned)Bg, 64, 0, stream>>>(segn, npg, u_e);
        graph_mean<<<(unsigned)Bg, 64, 0, stream>>>(xt1, npg, u_v);
        ASrc Au{};
        addSeg(Au, u_e, nullptr, 64);
        addSeg(Au, u_v, nullptr, 64);
        addSeg(Au, u_t, nullptr, 64);
        run_block(stream, Au, 192, Bg, convG[i], ut1, ut2, part, stats, zeroB);  // u_out in ut1

        // residuals
        size_t nE64 = (size_t)E * 64, nB64 = (size_t)Bg * 64;
        add2<<<(unsigned)((nN64 + 255) / 256), 256, 0, stream>>>(xc, xt1, (i == 0) ? x_t : xc, nN64);
        add2<<<(unsigned)((nE64 + 255) / 256), 256, 0, stream>>>(ec, et1, (i == 0) ? e_t : ec, nE64);
        add2<<<(unsigned)((nB64 + 255) / 256), 256, 0, stream>>>(uc, ut1, (i == 0) ? u_t : uc, nB64);

        e_in = ec; Ke = 64;
        u_in = uc; Ku = 64;
    }

    // ---- set2set over node features ----
    run_set2set(stream, xc, npg, Bg, WcX, bcX, hS, cS, qsx, gates, zeroB);

    // ---- e_node = seg_mean(ec, row, N); set2set over it ----
    size_t nN64 = (size_t)N * 64;
    kfill0<<<(unsigned)((nN64 + 255) / 256), 256, 0, stream>>>(segn, nN64);
    kfill0<<<(unsigned)((N + 255) / 256), 256, 0, stream>>>(cntn, (size_t)N);
    seg_scatter<<<(unsigned)(((size_t)E * 64 + 255) / 256), 256, 0, stream>>>(ec, row, E, segn, cntn);
    seg_div<<<(unsigned)((nN64 + 255) / 256), 256, 0, stream>>>(segn, cntn, N);
    run_set2set(stream, segn, npg, Bg, WcE, bcE, hS, cS, qse, gates, zeroB);

    // ---- post: relu(concat[x_pool, e_pool, uc] @ W + b) ----
    ASrc Ap{};
    addSeg(Ap, qsx, nullptr, 128);
    addSeg(Ap, qse, nullptr, 128);
    addSeg(Ap, uc, nullptr, 64);
    launch_gemm(stream, Ap, post.W, post.b, zeroB, pout, Bg, 320, 64, 1);

    // ---- final 64 -> 1 ----
    final_out<<<(unsigned)Bg, 64, 0, stream>>>(pout, fin.W, fin.b, (float*)d_out);
}
